// RSABlock_43954695307447
// MI455X (gfx1250) — matrix operand order repaired and checked by reading, not compiled, not run on hardware
//
#include <hip/hip_runtime.h>
#include <hip/hip_bf16.h>
#include <math.h>

// ---- problem constants (from reference) ----
#define Bn    2
#define Cc    32
#define Hh    256
#define Ww    256
#define DGn   8
#define TAPS  9
#define OMC   216        // 3 * DG * K offset/mask channels
#define OMCT  14         // ceil(216/16) channel tiles
#define KTOT  288        // Cc * TAPS
#define NEG_SLOPE 0.2f

typedef __attribute__((ext_vector_type(16))) __bf16 v16bf;
typedef __attribute__((ext_vector_type(8)))  float  v8f;

// workspace layout (bytes)
#define FEA_ELEMS   ((size_t)Bn * Cc * Hh * Ww)          // f32
#define POFF_ELEMS  ((size_t)OMCT * TAPS * 32 * 16)      // bf16 A-frags of w_off
#define PDCN_ELEMS  ((size_t)2 * TAPS * 32 * 16)         // bf16 A-frags of w_dcn
#define PC1_ELEMS   ((size_t)2 * TAPS * 32 * 16)         // bf16 A-frags of w_c1
#define OFF_POFF    (FEA_ELEMS * 4)
#define OFF_PDCN    (OFF_POFF + POFF_ELEMS * 2)
#define OFF_PC1     (OFF_PDCN + PDCN_ELEMS * 2)
#define OFF_PBOFF   (OFF_PC1 + PC1_ELEMS * 2)            // padded bias, 224 f32

// ---- WMMA fragment index helpers (contiguous k = 16*(lane/16) + e, 16-bit 16x32 A / 32x16 B) ----
__device__ __forceinline__ int a_kidx(int e, int lane) {
    return e + ((lane >= 16) ? 16 : 0);
}
__device__ __forceinline__ int b_kidx(int e, int lane) {
    return e + ((lane >= 16) ? 16 : 0);
}
__device__ __forceinline__ v8f wmma_bf16(v16bf a, v16bf b, v8f c) {
    return __builtin_amdgcn_wmma_f32_16x16x32_bf16(false, a, false, b, (short)0, c,
                                                   false, false);
}

// =====================================================================
// Prepack: f32 weights -> bf16 A-fragments in lane order; padded bias.
// =====================================================================
__global__ void pack_weights_kernel(const float* __restrict__ w_off,
                                    const float* __restrict__ w_dcn,
                                    const float* __restrict__ w_c1,
                                    const float* __restrict__ b_off,
                                    __bf16* __restrict__ p_off,
                                    __bf16* __restrict__ p_dcn,
                                    __bf16* __restrict__ p_c1,
                                    float* __restrict__ pb_off) {
    int i = blockIdx.x * 256 + threadIdx.x;
    const int NOFF = OMCT * TAPS * 512;  // 64512
    const int NDCN = 2 * TAPS * 512;     // 9216
    if (i < NOFF) {
        int e = i & 15, lane = (i >> 4) & 31, tt = i >> 9;
        int ct = tt / TAPS, t = tt % TAPS;
        int og = ct * 16 + (lane & 15);
        int c  = a_kidx(e, lane);
        float v = (og < OMC) ? w_off[((size_t)og * Cc + c) * TAPS + t] : 0.0f;
        p_off[i] = (__bf16)v;
    } else if (i < NOFF + NDCN) {
        int j = i - NOFF;
        int e = j & 15, lane = (j >> 4) & 31, tt = j >> 9;
        int mt = tt / TAPS, s = tt % TAPS;
        int m = mt * 16 + (lane & 15);
        p_dcn[j] = (__bf16)w_dcn[(size_t)m * KTOT + s * 32 + a_kidx(e, lane)];
    } else if (i < NOFF + 2 * NDCN) {
        int j = i - NOFF - NDCN;
        int e = j & 15, lane = (j >> 4) & 31, tt = j >> 9;
        int mt = tt / TAPS, t = tt % TAPS;
        int m = mt * 16 + (lane & 15);
        p_c1[j] = (__bf16)w_c1[((size_t)m * Cc + a_kidx(e, lane)) * TAPS + t];
    } else if (i < NOFF + 2 * NDCN + OMCT * 16) {
        int j = i - NOFF - 2 * NDCN;          // 0..223 padded bias
        pb_off[j] = (j < OMC) ? b_off[j] : 0.0f;
    }
}

// =====================================================================
// Kernel 1 (fused): conv_offset_mask (WMMA) -> LDS, deformable bilinear
// sampling + modulated contraction with w_dcn (WMMA), bias+leaky -> fea.
// One wave = one strip of 16 pixels along W; 2 waves per block.
// =====================================================================
__global__ __launch_bounds__(64) void dcn_fused_kernel(
    const float* __restrict__ x,
    const float* __restrict__ off,
    const v16bf* __restrict__ p_off,   // prepacked A-frags [ct*9+t][lane]
    const float* __restrict__ pb_off,  // padded bias [224]
    const v16bf* __restrict__ p_dcn,   // prepacked A-frags [mt*9+s][lane]
    const float* __restrict__ b_dcn,
    float* __restrict__ fea)
{
    __shared__ float lds_om[2][OMCT * 16][16];                      // 28 KB
    __shared__ __align__(32) __bf16 lds_in[2][3][18][Cc];           // 6.75 KB

    const int wave = threadIdx.x >> 5;
    const int lane = threadIdx.x & 31;
    const int strip = blockIdx.x * 2 + wave;
    const int b  = blockIdx.y;
    const int y  = strip >> 4;
    const int x0 = (strip & 15) << 4;
    const int n  = lane & 15;
    const int hi = (lane >= 16) ? 1 : 0;
    const int rbase = hi ? 8 : 0;

    // uniform per-block base pointers (SGPR) + 32-bit element offsets (GVS mode)
    const float* offb = off + ((size_t)b * Cc << 16);
    const float* xb   = x   + ((size_t)b * Cc << 16);
    float*       feab = fea + ((size_t)b * Cc << 16);

    // ---- stage 32ch x 3row x 18col patch as bf16, channel-innermost ----
    for (int i = lane; i < Cc * 3 * 18; i += 32) {
        int c = i / 54, r = (i / 18) % 3, col = i % 18;   // col fastest -> coalesced
        int iy = y + r - 1, ix = x0 + col - 1;
        float v = 0.0f;
        if (iy >= 0 && iy < Hh && ix >= 0 && ix < Ww)
            v = offb[(unsigned)((c << 16) + (iy << 8) + ix)];
        lds_in[wave][r][col][c] = (__bf16)v;
    }
    __syncthreads();

    // ---- phase 1: conv_offset_mask via WMMA ----
    for (int ct = 0; ct < OMCT; ++ct) {
        v8f acc = {};
#pragma unroll
        for (int t = 0; t < TAPS; ++t) {
            const int ty = t / 3, tx = t % 3;
            v16bf afr = p_off[(ct * TAPS + t) * 32 + lane];                 // 32B load
            v16bf bfr = *(const v16bf*)&lds_in[wave][ty][n + tx][hi * 16];  // 32B LDS
            acc = wmma_bf16(afr, bfr, acc);
        }
#pragma unroll 8
        for (int r = 0; r < 8; ++r) {
            int row = ct * 16 + rbase + r;               // rows 216..223 are dead pad
            float v = acc[r] + pb_off[row];
            if (row >= 2 * DGn * TAPS) v = 1.0f / (1.0f + __expf(-v));  // mask part
            lds_om[wave][row][n] = v;
        }
    }

    // ---- phase 2: deformable sampling + contraction (K=288, 9 steps) ----
    v8f o0 = {}, o1 = {};
    for (int s = 0; s < TAPS; ++s) {
        v16bf a0 = p_dcn[s * 32 + lane];
        v16bf a1 = p_dcn[(TAPS + s) * 32 + lane];
        v16bf bfr;
#pragma unroll 16
        for (int e = 0; e < 16; ++e) {
            int jb = s * 32 + b_kidx(e, lane);
            int ci = jb / TAPS, k = jb % TAPS, g = ci >> 2;
            float dy = lds_om[wave][g * TAPS + k][n];
            float dx = lds_om[wave][72 + g * TAPS + k][n];
            float mk = lds_om[wave][144 + g * TAPS + k][n];
            float py = (float)y + (float)(k / 3 - 1) + dy;
            float px = (float)(x0 + n) + (float)(k % 3 - 1) + dx;
            float fy = floorf(py), fx = floorf(px);
            int   iy = (int)fy,    ix = (int)fx;
            float ly = py - fy,    lx = px - fx;
            // clamp addresses, zero weights when OOB -> unconditional gathers
            int cy0 = min(max(iy, 0), Hh - 1),     cx0 = min(max(ix, 0), Ww - 1);
            int cy1 = min(max(iy + 1, 0), Hh - 1), cx1 = min(max(ix + 1, 0), Ww - 1);
            float vy0 = (iy >= 0 && iy < Hh) ? 1.0f : 0.0f;
            float vy1 = (iy + 1 >= 0 && iy + 1 < Hh) ? 1.0f : 0.0f;
            float vx0 = (ix >= 0 && ix < Ww) ? 1.0f : 0.0f;
            float vx1 = (ix + 1 >= 0 && ix + 1 < Ww) ? 1.0f : 0.0f;
            float w00 = (1.0f - ly) * (1.0f - lx) * vy0 * vx0;
            float w01 = (1.0f - ly) * lx          * vy0 * vx1;
            float w10 = ly * (1.0f - lx)          * vy1 * vx0;
            float w11 = ly * lx                   * vy1 * vx1;
            unsigned cbase = (unsigned)(ci << 16);
            float v = w00 * xb[cbase + (unsigned)((cy0 << 8) + cx0)]
                    + w01 * xb[cbase + (unsigned)((cy0 << 8) + cx1)]
                    + w10 * xb[cbase + (unsigned)((cy1 << 8) + cx0)]
                    + w11 * xb[cbase + (unsigned)((cy1 << 8) + cx1)];
            bfr[e] = (__bf16)(v * mk);
        }
        o0 = wmma_bf16(a0, bfr, o0);
        o1 = wmma_bf16(a1, bfr, o1);
    }

    // ---- bias + leaky relu -> fea ----
#pragma unroll 8
    for (int r = 0; r < 8; ++r) {
        {
            int o = rbase + r;
            float v = o0[r] + b_dcn[o];
            v = (v > 0.0f) ? v : NEG_SLOPE * v;
            feab[(unsigned)((o << 16) + (y << 8) + x0 + n)] = v;
        }
        {
            int o = 16 + rbase + r;
            float v = o1[r] + b_dcn[o];
            v = (v > 0.0f) ? v : NEG_SLOPE * v;
            feab[(unsigned)((o << 16) + (y << 8) + x0 + n)] = v;
        }
    }
}

// =====================================================================
// Kernel 2: conv1(fea) + bias + residual(x) -> out. 4 waves per block.
// =====================================================================
__global__ __launch_bounds__(128) void conv1_res_kernel(
    const float* __restrict__ fea,
    const v16bf* __restrict__ p_c1,   // prepacked A-frags [mt*9+t][lane]
    const float* __restrict__ b_c1,
    const float* __restrict__ x,
    float* __restrict__ out)
{
    __shared__ __align__(32) __bf16 lds_f[4][3][18][Cc];  // 13.5 KB

    const int wave = threadIdx.x >> 5;
    const int lane = threadIdx.x & 31;
    const int strip = blockIdx.x * 4 + wave;
    const int b  = blockIdx.y;
    const int y  = strip >> 4;
    const int x0 = (strip & 15) << 4;
    const int n  = lane & 15;
    const int hi = (lane >= 16) ? 1 : 0;
    const int rbase = hi ? 8 : 0;

    const float* feab = fea + ((size_t)b * Cc << 16);
    const float* xb   = x   + ((size_t)b * Cc << 16);
    float*       ob   = out + ((size_t)b * Cc << 16);

    for (int i = lane; i < Cc * 3 * 18; i += 32) {
        int c = i / 54, r = (i / 18) % 3, col = i % 18;
        int iy = y + r - 1, ix = x0 + col - 1;
        float v = 0.0f;
        if (iy >= 0 && iy < Hh && ix >= 0 && ix < Ww)
            v = feab[(unsigned)((c << 16) + (iy << 8) + ix)];
        lds_f[wave][r][col][c] = (__bf16)v;
    }
    __syncthreads();

    v8f o0 = {}, o1 = {};
#pragma unroll
    for (int t = 0; t < TAPS; ++t) {
        const int ty = t / 3, tx = t % 3;
        v16bf bfr = *(const v16bf*)&lds_f[wave][ty][n + tx][hi * 16];
        v16bf a0  = p_c1[t * 32 + lane];
        v16bf a1  = p_c1[(TAPS + t) * 32 + lane];
        o0 = wmma_bf16(a0, bfr, o0);
        o1 = wmma_bf16(a1, bfr, o1);
    }

#pragma unroll 8
    for (int r = 0; r < 8; ++r) {
        {
            int o = rbase + r;
            unsigned idx = (unsigned)((o << 16) + (y << 8) + x0 + n);
            ob[idx] = o0[r] + b_c1[o] + xb[idx];
        }
        {
            int o = 16 + rbase + r;
            unsigned idx = (unsigned)((o << 16) + (y << 8) + x0 + n);
            ob[idx] = o1[r] + b_c1[o] + xb[idx];
        }
    }
}

// =====================================================================
extern "C" void kernel_launch(void* const* d_in, const int* in_sizes, int n_in,
                              void* d_out, int out_size, void* d_ws, size_t ws_size,
                              hipStream_t stream) {
    const float* x     = (const float*)d_in[0];
    const float* off   = (const float*)d_in[1];
    const float* w_off = (const float*)d_in[2];
    const float* b_off = (const float*)d_in[3];
    const float* w_dcn = (const float*)d_in[4];
    const float* b_dcn = (const float*)d_in[5];
    const float* w_c1  = (const float*)d_in[6];
    const float* b_c1  = (const float*)d_in[7];
    float* out = (float*)d_out;

    char* ws = (char*)d_ws;
    float*  fea    = (float*)ws;                  // 16 MB
    __bf16* p_off  = (__bf16*)(ws + OFF_POFF);    // 126 KB
    __bf16* p_dcn  = (__bf16*)(ws + OFF_PDCN);    // 18 KB
    __bf16* p_c1   = (__bf16*)(ws + OFF_PC1);     // 18 KB
    float*  pb_off = (float*)(ws + OFF_PBOFF);    // 224 f32

    // 1) prepack weights into WMMA A-fragment order (bf16) + padded bias
    int total = (OMCT * TAPS + 4 * TAPS) * 512 + OMCT * 16;
    pack_weights_kernel<<<(total + 255) / 256, 256, 0, stream>>>(
        w_off, w_dcn, w_c1, b_off, p_off, p_dcn, p_c1, pb_off);

    // 2) fused offset-conv + deformable conv  (4096 strips/batch, 2 waves/block)
    dim3 g1(4096 / 2, Bn), blk1(64);
    dcn_fused_kernel<<<g1, blk1, 0, stream>>>(x, off, (const v16bf*)p_off, pb_off,
                                              (const v16bf*)p_dcn, b_dcn, fea);

    // 3) conv1 + bias + residual  (4 waves/block)
    dim3 g2(4096 / 4, Bn), blk2(128);
    conv1_res_kernel<<<g2, blk2, 0, stream>>>(fea, (const v16bf*)p_c1, b_c1, x, out);
}
